// AdaptiveEmbeddingI2T_65214783422652
// MI455X (gfx1250) — matrix operand order repaired and checked by reading, not compiled, not run on hardware
//
#include <hip/hip_runtime.h>
#include <math.h>

#define DD   1024
#define HH   128
#define BB   256
#define TT   72
#define RR   36
#define NBT  (BB * TT)          // 18432 BN samples per feature
#define BN_CHUNKS 32
#define BN_ROWS_PER_CHUNK (NBT / BN_CHUNKS)   // 576

typedef __attribute__((ext_vector_type(16))) __bf16 v16bf;
typedef __attribute__((ext_vector_type(8)))  float  v8f;

// ---------------------------------------------------------------------------
// WMMA helpers (gfx1250, wave32). EXEC must be all-ones: blockDim.x == 32,
// no divergent control flow in these kernels.
// ---------------------------------------------------------------------------
__device__ __forceinline__ v8f wmma_bf16(v16bf a, v16bf b, v8f c) {
  return __builtin_amdgcn_wmma_f32_16x16x32_bf16(
      /*neg_a=*/false, a, /*neg_b=*/false, b,
      /*c_mod=*/(short)0, c, /*reuse_a=*/false, /*reuse_b=*/false);
}

// A fragment: 16x32 bf16 from row-major MxK f32 matrix.
// Lane L (L&15)=M-row, hi=(L>>4): elems 0..15 -> K = k0+hi*16+e
// (contiguous k order, matching the B fragments).
__device__ __forceinline__ v16bf load_a_rowmajor(const float* M, int ld,
                                                 int row0, int k0) {
  const int lane = threadIdx.x & 31;
  const int m  = lane & 15;
  const int hi = lane >> 4;
  const float* p = M + (size_t)(row0 + m) * ld + k0 + hi * 16;
  v16bf a;
#pragma unroll
  for (int e = 0; e < 8; ++e) a[e] = (__bf16)p[e];
#pragma unroll
  for (int e = 0; e < 8; ++e) a[8 + e] = (__bf16)p[8 + e];
  return a;
}

// B fragment: 32x16 bf16, B[k][n] taken from an N-major f32 matrix
// (row-major N x K, contiguous in k).  Lane: n=(L&15), K = k0 + 16*hi + e.
__device__ __forceinline__ v16bf load_b_nmajor(const float* S, int ld,
                                               int col0, int k0) {
  const int lane = threadIdx.x & 31;
  const int n  = lane & 15;
  const int hi = lane >> 4;
  const float* p = S + (size_t)(col0 + n) * ld + k0 + hi * 16;
  v16bf b;
#pragma unroll
  for (int e = 0; e < 16; ++e) b[e] = (__bf16)p[e];
  return b;
}

// B fragment from a K-major f32 matrix (row-major K x N, contiguous in n).
__device__ __forceinline__ v16bf load_b_kmajor(const float* S, int ld,
                                               int col0, int k0) {
  const int lane = threadIdx.x & 31;
  const int n  = lane & 15;
  const int hi = lane >> 4;
  const float* p = S + (size_t)(k0 + hi * 16) * ld + col0 + n;
  v16bf b;
#pragma unroll
  for (int e = 0; e < 16; ++e) b[e] = (__bf16)p[(size_t)e * ld];
  return b;
}

// ---------------------------------------------------------------------------
// Block reduction (blockDim.x == 256)
// ---------------------------------------------------------------------------
__device__ __forceinline__ float block_reduce_sum_256(float v, float* sbuf) {
  const int t = threadIdx.x;
  sbuf[t] = v;
  __syncthreads();
#pragma unroll
  for (int s = 128; s > 0; s >>= 1) {
    if (t < s) sbuf[t] += sbuf[t + s];
    __syncthreads();
  }
  float r = sbuf[0];
  __syncthreads();
  return r;
}

// ---------------------------------------------------------------------------
// 1) img_repr = mean over 36 regions; rnorm = 1/(||img_repr|| + 1e-8)
//    one block per image, 256 threads, 4 features per thread
// ---------------------------------------------------------------------------
__global__ void k_img_mean(const float* __restrict__ img,
                           float* __restrict__ img_repr,
                           float* __restrict__ rnorm) {
  __shared__ float sbuf[256];
  const int i = blockIdx.x;
  const int t = threadIdx.x;
  const float* base = img + (size_t)i * RR * DD;
  float m[4] = {0.f, 0.f, 0.f, 0.f};
  for (int r = 0; r < RR; ++r) {
    const float* row = base + (size_t)r * DD;
#pragma unroll
    for (int c = 0; c < 4; ++c) m[c] += row[t + c * 256];
  }
  float ss = 0.f;
#pragma unroll
  for (int c = 0; c < 4; ++c) {
    m[c] *= (1.f / (float)RR);
    img_repr[(size_t)i * DD + t + c * 256] = m[c];
    ss += m[c] * m[c];
  }
  float tot = block_reduce_sum_256(ss, sbuf);
  if (t == 0) rnorm[i] = 1.f / (sqrtf(tot) + 1e-8f);
}

// ---------------------------------------------------------------------------
// 2a) BN partial stats: grid (4, 32), block 256
// ---------------------------------------------------------------------------
__global__ void k_bn_partial(const float* __restrict__ cap,
                             float* __restrict__ psum,
                             float* __restrict__ psq) {
  const int d = blockIdx.x * 256 + threadIdx.x;          // feature
  const int chunk = blockIdx.y;
  const int r0 = chunk * BN_ROWS_PER_CHUNK;
  float s = 0.f, q = 0.f;
  for (int r = r0; r < r0 + BN_ROWS_PER_CHUNK; ++r) {
    float v = cap[(size_t)r * DD + d];
    s += v;
    q += v * v;
  }
  psum[(size_t)chunk * DD + d] = s;
  psq [(size_t)chunk * DD + d] = q;
}

// ---------------------------------------------------------------------------
// 2b) BN final: mean, rstd.  grid 4, block 256
// ---------------------------------------------------------------------------
__global__ void k_bn_final(const float* __restrict__ psum,
                           const float* __restrict__ psq,
                           float* __restrict__ mean,
                           float* __restrict__ rstd) {
  const int d = blockIdx.x * 256 + threadIdx.x;
  float s = 0.f, q = 0.f;
  for (int c = 0; c < BN_CHUNKS; ++c) {
    s += psum[(size_t)c * DD + d];
    q += psq [(size_t)c * DD + d];
  }
  const float inv_n = 1.f / (float)NBT;
  float mu  = s * inv_n;
  float var = q * inv_n - mu * mu;
  mean[d] = mu;
  rstd[d] = 1.f / sqrtf(var + 1e-5f);
}

// ---------------------------------------------------------------------------
// 3) masked mean pooling of BN-normalized captions
//    pooled[j,d] = (sum_{t<len} cap - len*mean) * rstd / len
// ---------------------------------------------------------------------------
__global__ void k_pool(const float* __restrict__ cap,
                       const int* __restrict__ lens,
                       const float* __restrict__ mean,
                       const float* __restrict__ rstd,
                       float* __restrict__ pooled) {
  const int j = blockIdx.x;
  const int t = threadIdx.x;
  const int len = lens[j];
  const float flen = (float)len;
  const float* base = cap + (size_t)j * TT * DD;
#pragma unroll
  for (int c = 0; c < 4; ++c) {
    const int d = t + c * 256;
    float s = 0.f;
    for (int tt = 0; tt < len; ++tt) s += base[(size_t)tt * DD + d];
    pooled[(size_t)j * DD + d] = (s - flen * mean[d]) * rstd[d] / flen;
  }
}

// ---------------------------------------------------------------------------
// 4a) MLP layer 1: H = relu(img_repr @ W1 + b1).  grid (16, 8, 2), block 32.
//     z selects (Wg1,bg1) vs (Wb1,bb1).
// ---------------------------------------------------------------------------
__global__ void k_mlp1(const float* __restrict__ X,
                       const float* __restrict__ Wg1, const float* __restrict__ bg1,
                       const float* __restrict__ Wb1, const float* __restrict__ bb1,
                       float* __restrict__ Hbuf) {
  const int z  = blockIdx.z;
  const float* W    = z ? Wb1 : Wg1;
  const float* bias = z ? bb1 : bg1;
  const int tm = blockIdx.x * 16;
  const int tn = blockIdx.y * 16;
  const int lane = threadIdx.x & 31;
  const int n  = lane & 15;
  const int hi = lane >> 4;
  const float bv = bias[tn + n];
  v8f acc;
#pragma unroll
  for (int r = 0; r < 8; ++r) acc[r] = bv;
  for (int k0 = 0; k0 < DD; k0 += 32) {
    v16bf a = load_a_rowmajor(X, DD, tm, k0);
    v16bf b = load_b_kmajor(W, HH, tn, k0);
    acc = wmma_bf16(a, b, acc);
  }
  float* Hd = Hbuf + (size_t)z * BB * HH;
#pragma unroll
  for (int r = 0; r < 8; ++r)
    Hd[(size_t)(tm + r + hi * 8) * HH + tn + n] = fmaxf(acc[r], 0.f);
}

// ---------------------------------------------------------------------------
// 4b) MLP layer 2: out = H @ W2 + b2.  grid (16, 64, 2), block 32.
// ---------------------------------------------------------------------------
__global__ void k_mlp2(const float* __restrict__ Hbuf,
                       const float* __restrict__ Wg2, const float* __restrict__ bg2,
                       const float* __restrict__ Wb2, const float* __restrict__ bb2,
                       float* __restrict__ gammas, float* __restrict__ betas) {
  const int z  = blockIdx.z;
  const float* W    = z ? Wb2 : Wg2;
  const float* bias = z ? bb2 : bg2;
  const float* X    = Hbuf + (size_t)z * BB * HH;
  float* out        = z ? betas : gammas;
  const int tm = blockIdx.x * 16;
  const int tn = blockIdx.y * 16;
  const int lane = threadIdx.x & 31;
  const int n  = lane & 15;
  const int hi = lane >> 4;
  const float bv = bias[tn + n];
  v8f acc;
#pragma unroll
  for (int r = 0; r < 8; ++r) acc[r] = bv;
#pragma unroll
  for (int k0 = 0; k0 < HH; k0 += 32) {
    v16bf a = load_a_rowmajor(X, HH, tm, k0);
    v16bf b = load_b_kmajor(W, DD, tn, k0);
    acc = wmma_bf16(a, b, acc);
  }
#pragma unroll
  for (int r = 0; r < 8; ++r)
    out[(size_t)(tm + r + hi * 8) * DD + tn + n] = acc[r];
}

// ---------------------------------------------------------------------------
// 5) prep: U = img_n*(1+g), A = (1+g)^2, C2 = 2(1+g)b,
//          c_num = sum img_n*b, c_den = sum b^2
// ---------------------------------------------------------------------------
__global__ void k_prep(const float* __restrict__ img_repr,
                       const float* __restrict__ rnorm,
                       const float* __restrict__ gammas,
                       const float* __restrict__ betas,
                       float* __restrict__ U, float* __restrict__ Ac,
                       float* __restrict__ C2,
                       float* __restrict__ c_num, float* __restrict__ c_den) {
  __shared__ float sbuf[256];
  const int i = blockIdx.x;
  const int t = threadIdx.x;
  const float rn = rnorm[i];
  float pn = 0.f, pd = 0.f;
#pragma unroll
  for (int c = 0; c < 4; ++c) {
    const size_t idx = (size_t)i * DD + t + c * 256;
    const float g1  = 1.f + gammas[idx];
    const float b   = betas[idx];
    const float imn = img_repr[idx] * rn;
    U [idx] = imn * g1;
    Ac[idx] = g1 * g1;
    C2[idx] = 2.f * g1 * b;
    pn += imn * b;
    pd += b * b;
  }
  float tn = block_reduce_sum_256(pn, sbuf);
  float td = block_reduce_sum_256(pd, sbuf);
  if (t == 0) { c_num[i] = tn; c_den[i] = td; }
}

// ---------------------------------------------------------------------------
// 6) fused similarity: three shared-operand WMMA GEMMs + epilogue.
//    grid (16, 16), block 32.  out[j,i] = num / (sqrt(den) + 1e-8)
// ---------------------------------------------------------------------------
__global__ void k_sim(const float* __restrict__ U,
                      const float* __restrict__ Ac,
                      const float* __restrict__ C2,
                      const float* __restrict__ pooled,
                      const float* __restrict__ c_num,
                      const float* __restrict__ c_den,
                      float* __restrict__ out) {
  const int tm = blockIdx.x * 16;   // image rows i
  const int tn = blockIdx.y * 16;   // caption cols j
  const int lane = threadIdx.x & 31;
  const int n  = lane & 15;
  const int hi = lane >> 4;
  v8f accN = {}, accA = {}, accC = {};
  for (int k0 = 0; k0 < DD; k0 += 32) {
    v16bf au = load_a_rowmajor(U,  DD, tm, k0);
    v16bf aA = load_a_rowmajor(Ac, DD, tm, k0);
    v16bf aC = load_a_rowmajor(C2, DD, tm, k0);
    // pooled fragment + squared fragment (built in-register)
    const float* p = pooled + (size_t)(tn + n) * DD + k0 + hi * 16;
    v16bf bp, bq;
#pragma unroll
    for (int e = 0; e < 16; ++e) {
      float v = p[e];
      bp[e] = (__bf16)v;
      bq[e] = (__bf16)(v * v);
    }
    accN = wmma_bf16(au, bp, accN);
    accA = wmma_bf16(aA, bq, accA);
    accC = wmma_bf16(aC, bp, accC);
  }
#pragma unroll
  for (int r = 0; r < 8; ++r) {
    const int i = tm + r + hi * 8;
    const int j = tn + n;
    float num = accN[r] + c_num[i];
    float den = accA[r] + accC[r] + c_den[i];
    out[(size_t)j * BB + i] = num / (sqrtf(fmaxf(den, 0.f)) + 1e-8f);
  }
}

// ---------------------------------------------------------------------------
extern "C" void kernel_launch(void* const* d_in, const int* in_sizes, int n_in,
                              void* d_out, int out_size, void* d_ws, size_t ws_size,
                              hipStream_t stream) {
  const float* img_embed = (const float*)d_in[0];
  const float* cap_embed = (const float*)d_in[1];
  const int*   lens      = (const int*)  d_in[2];
  const float* Wg1 = (const float*)d_in[3];
  const float* bg1 = (const float*)d_in[4];
  const float* Wg2 = (const float*)d_in[5];
  const float* bg2 = (const float*)d_in[6];
  const float* Wb1 = (const float*)d_in[7];
  const float* bb1 = (const float*)d_in[8];
  const float* Wb2 = (const float*)d_in[9];
  const float* bb2 = (const float*)d_in[10];
  float* out = (float*)d_out;

  float* w = (float*)d_ws;
  float* img_repr = w; w += (size_t)BB * DD;
  float* rnorm    = w; w += BB;
  float* bn_s     = w; w += (size_t)BN_CHUNKS * DD;
  float* bn_q     = w; w += (size_t)BN_CHUNKS * DD;
  float* mean     = w; w += DD;
  float* rstd     = w; w += DD;
  float* pooled   = w; w += (size_t)BB * DD;
  float* Hbuf     = w; w += (size_t)2 * BB * HH;
  float* gammas   = w; w += (size_t)BB * DD;
  float* betas    = w; w += (size_t)BB * DD;
  float* U        = w; w += (size_t)BB * DD;
  float* Ac       = w; w += (size_t)BB * DD;
  float* C2       = w; w += (size_t)BB * DD;
  float* c_num    = w; w += BB;
  float* c_den    = w; w += BB;

  k_img_mean  <<<BB, 256, 0, stream>>>(img_embed, img_repr, rnorm);
  k_bn_partial<<<dim3(DD / 256, BN_CHUNKS), 256, 0, stream>>>(cap_embed, bn_s, bn_q);
  k_bn_final  <<<DD / 256, 256, 0, stream>>>(bn_s, bn_q, mean, rstd);
  k_pool      <<<BB, 256, 0, stream>>>(cap_embed, lens, mean, rstd, pooled);
  k_mlp1      <<<dim3(BB / 16, HH / 16, 2), 32, 0, stream>>>(img_repr, Wg1, bg1, Wb1, bb1, Hbuf);
  k_mlp2      <<<dim3(BB / 16, DD / 16, 2), 32, 0, stream>>>(Hbuf, Wg2, bg2, Wb2, bb2, gammas, betas);
  k_prep      <<<BB, 256, 0, stream>>>(img_repr, rnorm, gammas, betas, U, Ac, C2, c_num, c_den);
  k_sim       <<<dim3(BB / 16, BB / 16), 32, 0, stream>>>(U, Ac, C2, pooled, c_num, c_den, out);
}
